// DeepSets_83167746720066
// MI455X (gfx1250) — matrix operand order repaired and checked by reading, not compiled, not run on hardware
//
#include <hip/hip_runtime.h>

typedef __attribute__((ext_vector_type(16))) _Float16 v16h;
typedef __attribute__((ext_vector_type(8)))  _Float16 v8h;
typedef __attribute__((ext_vector_type(8)))  float    v8f;

#define FEAT 128
#define SEGS 256
#define WPM  16384          // f16 elements per packed 128x128 matrix
#define STR  136            // staging row stride in halves (128 + 8 pad -> conflict-free)

__device__ __forceinline__ float fast_tanh(float x) {
  float cx = fminf(fmaxf(x, -15.0f), 15.0f);
  float t  = __builtin_amdgcn_exp2f(cx * 2.885390081777927f);   // exp(2x)
  return (t - 1.0f) * __builtin_amdgcn_rcpf(t + 1.0f);
}

// ---------------------------------------------------------------------------
// Pack 4 weight matrices (gW1,gW2,lW1,lW2) fp32 -> f16 in WMMA-B per-lane
// order.  For B tile (kt,l2,n): lane needs halves h=0..15 holding
// W[32*kt + 8*l2 + (h&7) + 16*(h>>3)][n].  Stored split: h<8 in "lo" region, h>=8 in "hi"
// region (8 contiguous halves = one 16B LDS read each, bank-conflict free).
// Also zeroes the pooled-sum accumulator.
// ---------------------------------------------------------------------------
__global__ void prepack_zero(const float* __restrict__ gW1, const float* __restrict__ gW2,
                             const float* __restrict__ lW1, const float* __restrict__ lW2,
                             _Float16* __restrict__ wpack, float* __restrict__ pool) {
  int t   = blockIdx.x * 256 + threadIdx.x;     // 0 .. 65535
  int i   = t & (WPM - 1);
  int mtx = t >> 14;
  int h   = i & 15;
  int n   = (i >> 4) & 127;
  int l2  = (i >> 11) & 1;
  int kt  = (i >> 12) & 3;
  const float* W = (mtx == 0) ? gW1 : (mtx == 1) ? gW2 : (mtx == 2) ? lW1 : lW2;
  float v = W[(32 * kt + 8 * l2 + (h & 7) + 16 * (h >> 3)) * FEAT + n];
  int idx = (kt * 2 + l2) * 128 + n;
  wpack[mtx * WPM + (h >> 3) * 8192 + idx * 8 + (h & 7)] = (_Float16)v;
  if (t < SEGS * FEAT) pool[t] = 0.0f;
}

// ---------------------------------------------------------------------------
// One MLP branch: y = relu(x@W1 + b1)@W2  (+ epilogue per mode).
// mode 0: segment-sum y into pool (gamma branch, b2 folded in later)
// mode 1: out = tanh(y + b2 + pool[seg])   (lambda branch)
// 8 waves/block, each wave computes a 16x128 row tile with
// v_wmma_f32_16x16x32_f16 (64 WMMA per tile).
// ---------------------------------------------------------------------------
__launch_bounds__(256)
__global__ void mlp_pass(const float* __restrict__ x,
                         const int*   __restrict__ seg,
                         const _Float16* __restrict__ wpack,   // W1 pack, W2 pack follows
                         const float* __restrict__ b1,
                         const float* __restrict__ b2,
                         float* __restrict__ pool,
                         float* __restrict__ out,
                         int mode, int N, int nTiles, int nIter)
{
  __shared__ _Float16 wlds[2 * WPM];        // 64 KB: W1 + W2, pre-swizzled
  __shared__ _Float16 hlds[8][16 * STR];    // ~34 KB: per-wave f16 transpose staging

  { // cooperative weight copy (already in final layout)
    const uint4* src = (const uint4*)wpack;
    uint4* dst = (uint4*)wlds;
    #pragma unroll 4
    for (int i = threadIdx.x; i < (2 * WPM) / 8; i += 256) dst[i] = src[i];
  }
  __syncthreads();

  const int wave = threadIdx.x >> 5;
  const int lane = threadIdx.x & 31;
  const int l2   = lane >> 4;          // half-wave select
  const int m    = lane & 15;          // row (A) / column (B,C) within tile

  float bb1[8], bb2[8];
  #pragma unroll
  for (int nt = 0; nt < 8; ++nt) {
    bb1[nt] = b1[nt * 16 + m];
    bb2[nt] = (mode == 1) ? b2[nt * 16 + m] : 0.0f;
  }

  const v8f zero8 = {0.f, 0.f, 0.f, 0.f, 0.f, 0.f, 0.f, 0.f};
  _Float16* hst = &hlds[wave][0];

  for (int it = 0; it < nIter; ++it) {
    int tile   = (it * gridDim.x + blockIdx.x) * 8 + wave;
    bool valid = tile < nTiles;
    int tbase  = valid ? tile * 16 : 0;

    // ---------------- GEMM1: h = x_tile @ W1 ----------------
    v8f acc[8];
    #pragma unroll
    for (int nt = 0; nt < 8; ++nt) acc[nt] = zero8;

    int rowl = tbase + m; if (rowl > N - 1) rowl = N - 1;
    const float* xr = x + (size_t)rowl * FEAT;

    #pragma unroll
    for (int kt = 0; kt < 4; ++kt) {
      // A-matrix 16x32 f16 layout: lane<16 holds K = 32kt+{0..7,16..23},
      // lane>=16 holds K = 32kt+{8..15,24..31}.
      int c0 = 32 * kt + 8 * l2;
      float4 f0 = *(const float4*)(xr + c0);
      float4 f1 = *(const float4*)(xr + c0 + 4);
      float4 f2 = *(const float4*)(xr + c0 + 16);
      float4 f3 = *(const float4*)(xr + c0 + 20);
      v16h a;
      a[0]=(_Float16)f0.x;  a[1]=(_Float16)f0.y;  a[2]=(_Float16)f0.z;  a[3]=(_Float16)f0.w;
      a[4]=(_Float16)f1.x;  a[5]=(_Float16)f1.y;  a[6]=(_Float16)f1.z;  a[7]=(_Float16)f1.w;
      a[8]=(_Float16)f2.x;  a[9]=(_Float16)f2.y;  a[10]=(_Float16)f2.z; a[11]=(_Float16)f2.w;
      a[12]=(_Float16)f3.x; a[13]=(_Float16)f3.y; a[14]=(_Float16)f3.z; a[15]=(_Float16)f3.w;
      #pragma unroll
      for (int nt = 0; nt < 8; ++nt) {
        int idx = (kt * 2 + l2) * 128 + nt * 16 + m;
        v16h b;
        ((v8h*)&b)[0] = *(const v8h*)&wlds[idx * 8];
        ((v8h*)&b)[1] = *(const v8h*)&wlds[8192 + idx * 8];
        acc[nt] = __builtin_amdgcn_wmma_f32_16x16x32_f16(false, a, false, b,
                                                         (short)0, acc[nt], false, false);
      }
    }

    // bias + relu, stage f16 (C layout -> row-major LDS, padded stride)
    #pragma unroll
    for (int nt = 0; nt < 8; ++nt) {
      #pragma unroll
      for (int r = 0; r < 8; ++r) {
        float v = fmaxf(acc[nt][r] + bb1[nt], 0.0f);
        hst[(r + 8 * l2) * STR + nt * 16 + m] = (_Float16)v;
      }
    }
    __syncthreads();

    // ---------------- GEMM2: y = h @ W2 ----------------
    v8f acc2[8];
    #pragma unroll
    for (int nt = 0; nt < 8; ++nt) acc2[nt] = zero8;

    const _Float16* hr = hst + m * STR;
    #pragma unroll
    for (int kt = 0; kt < 4; ++kt) {
      int c0 = 32 * kt + 8 * l2;
      v16h a;
      ((v8h*)&a)[0] = *(const v8h*)(hr + c0);
      ((v8h*)&a)[1] = *(const v8h*)(hr + c0 + 16);
      #pragma unroll
      for (int nt = 0; nt < 8; ++nt) {
        int idx = (kt * 2 + l2) * 128 + nt * 16 + m;
        v16h b;
        ((v8h*)&b)[0] = *(const v8h*)&wlds[WPM + idx * 8];
        ((v8h*)&b)[1] = *(const v8h*)&wlds[WPM + 8192 + idx * 8];
        acc2[nt] = __builtin_amdgcn_wmma_f32_16x16x32_f16(false, a, false, b,
                                                          (short)0, acc2[nt], false, false);
      }
    }

    // ---------------- epilogue ----------------
    if (mode == 0) {
      if (valid) {
        int last = (tbase + 15 < N) ? tbase + 15 : N - 1;
        int s0 = seg[tbase], s1 = seg[last];
        if (s0 == s1 && tbase + 15 < N) {
          // uniform-segment tile: reduce 16 rows first, 128 atomics/tile
          #pragma unroll
          for (int nt = 0; nt < 8; ++nt) {
            float sum = 0.0f;
            #pragma unroll
            for (int r = 0; r < 8; ++r) sum += acc2[nt][r];
            sum += __shfl_xor(sum, 16, 32);
            if (l2 == 0) atomicAdd(&pool[s0 * FEAT + nt * 16 + m], sum);
          }
        } else {
          #pragma unroll
          for (int r = 0; r < 8; ++r) {
            int row = tbase + r + 8 * l2;
            if (row < N) {
              int s = seg[row];
              #pragma unroll
              for (int nt = 0; nt < 8; ++nt)
                atomicAdd(&pool[s * FEAT + nt * 16 + m], acc2[nt][r]);
            }
          }
        }
      }
    } else {
      if (valid) {
        #pragma unroll
        for (int r = 0; r < 8; ++r) {
          int row = tbase + r + 8 * l2;
          if (row < N) {
            const float* pr = pool + (size_t)seg[row] * FEAT;
            #pragma unroll
            for (int nt = 0; nt < 8; ++nt) {
              int n = nt * 16 + m;
              out[(size_t)row * FEAT + n] = fast_tanh(acc2[nt][r] + bb2[nt] + pr[n]);
            }
          }
        }
      }
    }
  }
}

// ---------------------------------------------------------------------------
// pooled[s] = pooled_sum[s] / max(count_s,1) + gb2 ; counts by binary search
// over sorted segment_ids (no count atomics anywhere).
// ---------------------------------------------------------------------------
__global__ void finalize_pool(const int* __restrict__ seg, const float* __restrict__ gb2,
                              float* __restrict__ pool, int N) {
  __shared__ float invc;
  int s = blockIdx.x;
  if (threadIdx.x == 0) {
    int lo = 0, hi = N;
    while (lo < hi) { int mid = (lo + hi) >> 1; if (seg[mid] < s) lo = mid + 1; else hi = mid; }
    int a = lo;
    lo = 0; hi = N;
    while (lo < hi) { int mid = (lo + hi) >> 1; if (seg[mid] < s + 1) lo = mid + 1; else hi = mid; }
    invc = 1.0f / fmaxf((float)(lo - a), 1.0f);
  }
  __syncthreads();
  int t = threadIdx.x;   // 128 threads
  pool[s * FEAT + t] = pool[s * FEAT + t] * invc + gb2[t];
}

extern "C" void kernel_launch(void* const* d_in, const int* in_sizes, int n_in,
                              void* d_out, int out_size, void* d_ws, size_t ws_size,
                              hipStream_t stream) {
  const float* x   = (const float*)d_in[0];
  const int*   seg = (const int*)  d_in[1];
  const float* lW1 = (const float*)d_in[2];
  const float* lb1 = (const float*)d_in[3];
  const float* lW2 = (const float*)d_in[4];
  const float* lb2 = (const float*)d_in[5];
  const float* gW1 = (const float*)d_in[6];
  const float* gb1 = (const float*)d_in[7];
  const float* gW2 = (const float*)d_in[8];
  const float* gb2 = (const float*)d_in[9];
  float* out = (float*)d_out;

  int N = in_sizes[0] / FEAT;
  _Float16* wpack = (_Float16*)d_ws;                        // 4 * 32 KB
  float* pool = (float*)((char*)d_ws + 4 * WPM * sizeof(_Float16)); // 128 KB

  int nTiles = (N + 15) / 16;
  int G = 640;
  int nIter = (nTiles + G * 8 - 1) / (G * 8);

  prepack_zero<<<256, 256, 0, stream>>>(gW1, gW2, lW1, lW2, wpack, pool);
  // gamma branch: segment-sum accumulate
  mlp_pass<<<G, 256, 0, stream>>>(x, seg, wpack, gb1, nullptr, pool, nullptr,
                                  0, N, nTiles, nIter);
  finalize_pool<<<SEGS, FEAT, 0, stream>>>(seg, gb2, pool, N);
  // lambda branch + broadcast-add + tanh
  mlp_pass<<<G, 256, 0, stream>>>(x, seg, wpack + 2 * WPM, lb1, lb2, pool, out,
                                  1, N, nTiles, nIter);
}
